// Attention_B_75505525064252
// MI455X (gfx1250) — matrix operand order repaired and checked by reading, not compiled, not run on hardware
//
#include <hip/hip_runtime.h>

// ---------- CDNA5 vector types ----------
typedef __bf16 v16bf  __attribute__((ext_vector_type(16)));
typedef __bf16 bf16x8 __attribute__((ext_vector_type(8)));
typedef float  v8f    __attribute__((ext_vector_type(8)));
typedef unsigned u32x4 __attribute__((ext_vector_type(4)));
typedef unsigned u32x8 __attribute__((ext_vector_type(8)));

#define FLT_BIG 3.402823466e38f

__device__ __forceinline__ __bf16 f2bf(float f) {
    unsigned u = __builtin_bit_cast(unsigned, f);
    u += 0x7FFFu + ((u >> 16) & 1u);          // round-to-nearest-even
    unsigned short s = (unsigned short)(u >> 16);
    return __builtin_bit_cast(__bf16, s);
}

__device__ __forceinline__ v16bf mk16(bf16x8 lo, bf16x8 hi) {
    v16bf r;
#pragma unroll
    for (int i = 0; i < 8; i++) { r[i] = lo[i]; r[i + 8] = hi[i]; }
    return r;
}

__device__ __forceinline__ v8f zero8() {
    v8f z;
#pragma unroll
    for (int i = 0; i < 8; i++) z[i] = 0.f;
    return z;
}

// Issue a TDM 2D tile load (bf16, tile 32(K) x 128(rows)) into LDS.
// D# group0/group1 packed per CDNA5 ISA.
// Must be executed by a single wave; tracked by TENSORcnt.
__device__ __forceinline__ void tdm_load_tile_bf16(
    unsigned lds_addr, unsigned long long gaddr, unsigned k_stride_elems)
{
    u32x4 g0;
    g0[0] = 1u;                                   // count=1, user mode
    g0[1] = lds_addr;                             // LDS byte address
    g0[2] = (unsigned)gaddr;                      // global addr [31:0]
    g0[3] = (unsigned)((gaddr >> 32) & 0x1FFFFFFu) | 0x80000000u; // [56:32] | type=2

    u32x8 g1;
    g1[0] = (1u << 16);                           // wg_mask=0, data_size=1 (2B)
    g1[1] = 0xFFFF0000u;                          // tensor_dim0 lo16 (huge, no OOB)
    g1[2] = 0xFFFF7FFFu;                          // dim0 hi16 | dim1 lo16
    g1[3] = (32u << 16) | 0x7FFFu;                // tile_dim0=32 | dim1 hi16
    g1[4] = 128u;                                 // tile_dim1=128, tile_dim2=0
    g1[5] = k_stride_elems;                       // tensor_dim0_stride lo32
    g1[6] = 0u;                                   // stride hi16 | dim1_stride lo16
    g1[7] = 0u;                                   // dim1_stride hi32
    asm volatile("tensor_load_to_lds %0, %1" :: "s"(g0), "s"(g1) : "memory");
}

// =====================================================================
// Generic bf16-WMMA GEMM: C[M,N] = A[M,K] * B[K,N] + bias[N]
// Block tile 128x128, K-step 32, 256 threads = 8 waves (2M x 4N),
// each wave: 4x2 v_wmma_f32_16x16x32_bf16 tiles. Double-buffered LDS.
// A_BF16: A staged via TDM tensor_load_to_lds (raw bf16 copy).
// else:   A fp32, register-staged + converted to bf16.
// =====================================================================
template <bool A_BF16, bool C_FP32>
__global__ __launch_bounds__(256) void gemm_bf16_wmma(
    const void* __restrict__ Ap, const float* __restrict__ B,
    const float* __restrict__ bias, void* __restrict__ Cp,
    int M, int N, int K)
{
    __shared__ __bf16 As[2][128][32];   // 16 KB
    __shared__ __bf16 Bt[2][128][32];   // 16 KB, transposed [n][k]

    const int tid  = threadIdx.x;
    const int wave = tid >> 5, lane = tid & 31;
    const int hf   = lane >> 4, l15 = lane & 15;
    const int wm   = wave >> 2, wn  = wave & 3;
    const int bm   = blockIdx.y * 128, bn = blockIdx.x * 128;

    v8f acc[4][2];
#pragma unroll
    for (int i = 0; i < 4; i++)
#pragma unroll
        for (int j = 0; j < 2; j++) acc[i][j] = zero8();

    const int nk = K >> 5;

    float4 aR[4];      // fp32-A staging registers
    float4 bR[4];      // B staging registers

    // ---------------- prologue: fetch tile 0 ----------------
    if constexpr (A_BF16) {
        if (wave == 0)
            tdm_load_tile_bf16((unsigned)(size_t)&As[0][0][0],
                               (unsigned long long)(size_t)Ap + ((size_t)bm * K) * 2,
                               (unsigned)K);
    } else {
        const float* Af = (const float*)Ap;
#pragma unroll
        for (int i = 0; i < 4; i++) {
            int e = tid + 256 * i;
            int row = e >> 3, c4 = (e & 7) * 4;
            aR[i] = *(const float4*)(Af + (size_t)(bm + row) * K + c4);
        }
#pragma unroll
        for (int i = 0; i < 4; i++) {
            int e = tid + 256 * i;
            int row = e >> 3, c4 = (e & 7) * 4;
            As[0][row][c4 + 0] = f2bf(aR[i].x); As[0][row][c4 + 1] = f2bf(aR[i].y);
            As[0][row][c4 + 2] = f2bf(aR[i].z); As[0][row][c4 + 3] = f2bf(aR[i].w);
        }
    }
#pragma unroll
    for (int i = 0; i < 4; i++) {
        int e = tid + 256 * i;
        int k = e >> 5, n4 = (e & 31) * 4;
        bR[i] = *(const float4*)(B + (size_t)k * N + bn + n4);
    }
#pragma unroll
    for (int i = 0; i < 4; i++) {
        int e = tid + 256 * i;
        int k = e >> 5, n4 = (e & 31) * 4;
        Bt[0][n4 + 0][k] = f2bf(bR[i].x); Bt[0][n4 + 1][k] = f2bf(bR[i].y);
        Bt[0][n4 + 2][k] = f2bf(bR[i].z); Bt[0][n4 + 3][k] = f2bf(bR[i].w);
    }

    // ---------------- main loop, one barrier per K-step ----------------
    for (int kt = 0; kt < nk; ++kt) {
        const int cur = kt & 1, nxt = cur ^ 1;
        const int k0n = (kt + 1) << 5;
        const bool more = (kt + 1) < nk;

        if constexpr (A_BF16) {
            if (wave == 0) __builtin_amdgcn_s_wait_tensorcnt(0); // tile kt in LDS
        }
        __syncthreads();

        // issue next tile's fetches (overlap with WMMA below)
        if (more) {
            if constexpr (A_BF16) {
                if (wave == 0)
                    tdm_load_tile_bf16((unsigned)(size_t)&As[nxt][0][0],
                                       (unsigned long long)(size_t)Ap +
                                           ((size_t)bm * K + k0n) * 2,
                                       (unsigned)K);
            } else {
                const float* Af = (const float*)Ap;
#pragma unroll
                for (int i = 0; i < 4; i++) {
                    int e = tid + 256 * i;
                    int row = e >> 3, c4 = (e & 7) * 4;
                    aR[i] = *(const float4*)(Af + (size_t)(bm + row) * K + k0n + c4);
                }
            }
#pragma unroll
            for (int i = 0; i < 4; i++) {
                int e = tid + 256 * i;
                int k = e >> 5, n4 = (e & 31) * 4;
                bR[i] = *(const float4*)(B + (size_t)(k0n + k) * N + bn + n4);
            }
        }

        // ---- compute on current buffer ----
        v16bf bfrag[2];
#pragma unroll
        for (int j = 0; j < 2; j++) {
            int n = wn * 32 + j * 16 + l15;
            bfrag[j] = mk16(*(const bf16x8*)&Bt[cur][n][8 * hf], *(const bf16x8*)&Bt[cur][n][16 + 8 * hf]);
        }
#pragma unroll
        for (int i = 0; i < 4; i++) {
            int row = wm * 64 + i * 16 + l15;
            bf16x8 lo = *(const bf16x8*)&As[cur][row][8 * hf];
            bf16x8 hi = *(const bf16x8*)&As[cur][row][16 + 8 * hf];
            v16bf a = mk16(lo, hi);
#pragma unroll
            for (int j = 0; j < 2; j++)
                acc[i][j] = __builtin_amdgcn_wmma_f32_16x16x32_bf16(
                    false, a, false, bfrag[j], (short)0, acc[i][j], false, false);
        }

        // ---- store next tile into the other buffer ----
        if (more) {
            if constexpr (!A_BF16) {
#pragma unroll
                for (int i = 0; i < 4; i++) {
                    int e = tid + 256 * i;
                    int row = e >> 3, c4 = (e & 7) * 4;
                    As[nxt][row][c4 + 0] = f2bf(aR[i].x);
                    As[nxt][row][c4 + 1] = f2bf(aR[i].y);
                    As[nxt][row][c4 + 2] = f2bf(aR[i].z);
                    As[nxt][row][c4 + 3] = f2bf(aR[i].w);
                }
            }
#pragma unroll
            for (int i = 0; i < 4; i++) {
                int e = tid + 256 * i;
                int k = e >> 5, n4 = (e & 31) * 4;
                Bt[nxt][n4 + 0][k] = f2bf(bR[i].x);
                Bt[nxt][n4 + 1][k] = f2bf(bR[i].y);
                Bt[nxt][n4 + 2][k] = f2bf(bR[i].z);
                Bt[nxt][n4 + 3][k] = f2bf(bR[i].w);
            }
        }
    }

    // ---- epilogue: C/D layout row = r + 8*hf, col = l15 ----
#pragma unroll
    for (int i = 0; i < 4; i++)
#pragma unroll
        for (int j = 0; j < 2; j++)
#pragma unroll
            for (int r = 0; r < 8; r++) {
                int row = bm + wm * 64 + i * 16 + r + 8 * hf;
                int col = bn + wn * 32 + j * 16 + l15;
                float v = acc[i][j][r] + bias[col];
                if constexpr (C_FP32)
                    ((float*)Cp)[(size_t)row * N + col] = v;
                else
                    ((__bf16*)Cp)[(size_t)row * N + col] = f2bf(v);
            }
}

// =====================================================================
// Flash attention (causal), per-(b,h) 64-query tiles.
// qkv: bf16 [B*S][3E] workspace (Q|K|V column blocks of 1024).
// out: bf16 [B*S][E] merged-head layout.
// 128 threads = 4 waves; wave owns 16 query rows; key chunks of 32.
// =====================================================================
__global__ __launch_bounds__(128) void attn_causal_wmma(
    const __bf16* __restrict__ qkv, __bf16* __restrict__ outb)
{
    __shared__ __bf16 VT[64][32];        // V chunk transposed [d][key]
    __shared__ __bf16 Pb[4][16][32];     // per-wave P staging

    const int tid  = threadIdx.x;
    const int wave = tid >> 5, lane = tid & 31;
    const int hf   = lane >> 4, l15 = lane & 15;
    const int b    = blockIdx.y >> 4, h = blockIdx.y & 15;
    const int qbase = blockIdx.x * 64;
    const int qw    = qbase + wave * 16;
    const __bf16* base = qkv + (size_t)b * 1024 * 3072;

    // Q fragments (A-matrix lane layout) straight from global
    v16bf qfrag[2];
    {
        const __bf16* qrow = base + (size_t)(qw + l15) * 3072 + h * 64;
#pragma unroll
        for (int ds = 0; ds < 2; ++ds) {
            bf16x8 lo = *(const bf16x8*)(qrow + ds * 32 + 8 * hf);
            bf16x8 hi = *(const bf16x8*)(qrow + ds * 32 + 16 + 8 * hf);
            qfrag[ds] = mk16(lo, hi);
        }
    }

    float m[8], l[8];
    v8f o[4];
#pragma unroll
    for (int r = 0; r < 8; r++) { m[r] = -FLT_BIG; l[r] = 0.f; }
#pragma unroll
    for (int nt = 0; nt < 4; nt++) o[nt] = zero8();

    const float scale = 0.125f;   // 1/sqrt(64)

    for (int kc = 0; kc < qbase + 64; kc += 32) {
        // ---- stage V chunk (32 keys x 64 d) transposed into LDS ----
        {
            int key = tid >> 2, d0 = (tid & 3) * 16;
            const __bf16* vrow = base + (size_t)(kc + key) * 3072 + 2048 + h * 64 + d0;
            bf16x8 v0 = *(const bf16x8*)vrow;
            bf16x8 v1 = *(const bf16x8*)(vrow + 8);
#pragma unroll
            for (int i = 0; i < 8; i++) { VT[d0 + i][key] = v0[i]; VT[d0 + 8 + i][key] = v1[i]; }
        }
        __syncthreads();

        // ---- S = Q * K^T over D=64 ----
        v8f s[2];
#pragma unroll
        for (int t = 0; t < 2; t++) {
            v8f sa = zero8();
            int key = kc + t * 16 + l15;
            const __bf16* krow = base + (size_t)key * 3072 + 1024 + h * 64;
#pragma unroll
            for (int ds = 0; ds < 2; ++ds) {
                v16bf kf = mk16(*(const bf16x8*)(krow + ds * 32 + 8 * hf), *(const bf16x8*)(krow + ds * 32 + 16 + 8 * hf));
                sa = __builtin_amdgcn_wmma_f32_16x16x32_bf16(
                        false, qfrag[ds], false, kf, (short)0, sa, false, false);
            }
            s[t] = sa;
        }

        // ---- causal mask + online softmax ----
#pragma unroll
        for (int r = 0; r < 8; r++) {
            int qi = qw + r + 8 * hf;
#pragma unroll
            for (int t = 0; t < 2; t++) {
                int ki = kc + t * 16 + l15;
                float sv = s[t][r] * scale;
                s[t][r] = (ki <= qi) ? sv : -FLT_BIG;
            }
            float mx = fmaxf(s[0][r], s[1][r]);
#pragma unroll
            for (int off = 8; off >= 1; off >>= 1) mx = fmaxf(mx, __shfl_xor(mx, off, 32));
            float mn    = fmaxf(m[r], mx);
            float alpha = __expf(m[r] - mn);
            float p0    = __expf(s[0][r] - mn);
            float p1    = __expf(s[1][r] - mn);
            s[0][r] = p0; s[1][r] = p1;
            float rs = p0 + p1;
#pragma unroll
            for (int off = 8; off >= 1; off >>= 1) rs += __shfl_xor(rs, off, 32);
            l[r] = l[r] * alpha + rs;
            m[r] = mn;
#pragma unroll
            for (int nt = 0; nt < 4; nt++) o[nt][r] *= alpha;
        }

        // ---- P: C-layout -> LDS -> A-layout fragment ----
#pragma unroll
        for (int r = 0; r < 8; r++) {
            Pb[wave][r + 8 * hf][l15]      = f2bf(s[0][r]);
            Pb[wave][r + 8 * hf][16 + l15] = f2bf(s[1][r]);
        }
        __syncthreads();

        // ---- O += P * V ----
        {
            bf16x8 lo = *(const bf16x8*)&Pb[wave][l15][8 * hf];
            bf16x8 hi = *(const bf16x8*)&Pb[wave][l15][16 + 8 * hf];
            v16bf pf = mk16(lo, hi);
#pragma unroll
            for (int nt = 0; nt < 4; nt++) {
                v16bf vf = mk16(*(const bf16x8*)&VT[nt * 16 + l15][8 * hf], *(const bf16x8*)&VT[nt * 16 + l15][16 + 8 * hf]);
                o[nt] = __builtin_amdgcn_wmma_f32_16x16x32_bf16(
                            false, pf, false, vf, (short)0, o[nt], false, false);
            }
        }
        __syncthreads();
    }

    // ---- normalize and store merged-head bf16 [B*S][E] ----
#pragma unroll
    for (int nt = 0; nt < 4; nt++)
#pragma unroll
        for (int r = 0; r < 8; r++) {
            int row = qw + r + 8 * hf;
            int col = h * 64 + nt * 16 + l15;
            outb[(size_t)(b * 1024 + row) * 1024 + col] = f2bf(o[nt][r] / l[r]);
        }
}

// =====================================================================
// Host-side launch
// =====================================================================
extern "C" void kernel_launch(void* const* d_in, const int* in_sizes, int n_in,
                              void* d_out, int out_size, void* d_ws, size_t ws_size,
                              hipStream_t stream) {
    const float* x      = (const float*)d_in[0];   // [4,1024,1024]
    const float* W_attn = (const float*)d_in[1];   // [1024,3072]
    const float* b_attn = (const float*)d_in[2];   // [3072]
    const float* W_proj = (const float*)d_in[3];   // [1024,1024]
    const float* b_proj = (const float*)d_in[4];   // [1024]
    float* out = (float*)d_out;                    // [4,1024,1024]

    const int M = 4096, E = 1024, N3 = 3072;
    __bf16* qkv      = (__bf16*)d_ws;                                   // 4096x3072 bf16
    __bf16* attn_out = (__bf16*)((char*)d_ws + (size_t)M * N3 * 2);     // 4096x1024 bf16

    // 1) qkv = x @ W_attn + b_attn  (fp32 in, bf16 out)
    gemm_bf16_wmma<false, false><<<dim3(N3 / 128, M / 128), 256, 0, stream>>>(
        x, W_attn, b_attn, qkv, M, N3, E);

    // 2) causal flash attention per (b,h), 64-query tiles
    attn_causal_wmma<<<dim3(16, 64), 128, 0, stream>>>(qkv, attn_out);

    // 3) out = attn_out @ W_proj + b_proj  (bf16 A staged via TDM, fp32 out)
    gemm_bf16_wmma<true, true><<<dim3(E / 128, M / 128), 256, 0, stream>>>(
        attn_out, W_proj, b_proj, out, M, E, E);
}
